// CausalSelfAttention_41197326303385
// MI455X (gfx1250) — compile-verified, matrix operand order checked by reading, not run on hardware
//
#include <hip/hip_runtime.h>
#include <hip/hip_bf16.h>

#define B_   4
#define T_   2048
#define D_   1024
#define H_   16
#define HD_  64
#define BH_  (B_ * H_)
#define M_   (B_ * T_)
#define N_QKV (3 * D_)

#ifndef __has_builtin
#define __has_builtin(x) 0
#endif

#if __has_builtin(__builtin_amdgcn_global_load_async_to_lds_b128)
#define HAVE_ASYNC_COPY 1
#else
#define HAVE_ASYNC_COPY 0
#endif

#if __has_builtin(__builtin_amdgcn_s_wait_asynccnt)
#define WAIT_ASYNC(n) __builtin_amdgcn_s_wait_asynccnt(n)
#else
#define WAIT_ASYNC(n) asm volatile("s_wait_asynccnt %0" ::"n"(n) : "memory")
#endif

typedef __attribute__((ext_vector_type(16))) __bf16 bf16x16;
typedef __attribute__((ext_vector_type(8)))  float  f32x8;
typedef int v4i_vs __attribute__((vector_size(16)));   // matches builtin param type

__device__ __forceinline__ unsigned short f32_bf16(float f) {
    unsigned u = __builtin_bit_cast(unsigned, f);
    u += 0x7FFFu + ((u >> 16) & 1u);   // round-to-nearest-even
    return (unsigned short)(u >> 16);
}

// Build a 16-element bf16 fragment from two contiguous 16-byte chunks.
__device__ __forceinline__ bf16x16 load_frag(const unsigned short* p0,
                                             const unsigned short* p1) {
    bf16x16 f;
    ((uint4*)&f)[0] = *(const uint4*)p0;
    ((uint4*)&f)[1] = *(const uint4*)p1;
    return f;
}

__device__ __forceinline__ f32x8 wmma_bf16(bf16x16 a, bf16x16 b, f32x8 c) {
    return __builtin_amdgcn_wmma_f32_16x16x32_bf16(false, a, false, b,
                                                   (short)0, c, false, false);
}

// 16-byte global -> LDS copy: async (ASYNCcnt) when available, else via VGPRs.
__device__ __forceinline__ void cp16_g2l(const unsigned short* g, unsigned short* l) {
#if HAVE_ASYNC_COPY
    __builtin_amdgcn_global_load_async_to_lds_b128(
        (__attribute__((address_space(1))) v4i_vs*)g,
        (__attribute__((address_space(3))) v4i_vs*)l, 0, 0);
#else
    *(uint4*)l = *(const uint4*)g;
#endif
}

// ---------------------------------------------------------------- cast fp32->bf16
__global__ void k_cast(const float* __restrict__ in,
                       unsigned short* __restrict__ out, int n) {
    int i = (blockIdx.x * 256 + threadIdx.x) << 2;
    if (i < n) {
        float4 v = *(const float4*)(in + i);
        union { unsigned short us[4]; uint2 u; } r;
        r.us[0] = f32_bf16(v.x);
        r.us[1] = f32_bf16(v.y);
        r.us[2] = f32_bf16(v.z);
        r.us[3] = f32_bf16(v.w);
        *(uint2*)(out + i) = r.u;
    }
}

// ------------------------------------------- transpose+cast: w[K][N] -> out[N][K]
__global__ __launch_bounds__(256)
void k_transpose_cast(const float* __restrict__ w,
                      unsigned short* __restrict__ out, int K, int N) {
    __shared__ __align__(16) unsigned short tile[32][33];
    int nb = N >> 5;
    int n0 = (blockIdx.x % nb) << 5;
    int k0 = (blockIdx.x / nb) << 5;
    int lx = threadIdx.x & 31, ly = threadIdx.x >> 5;
#pragma unroll
    for (int i = 0; i < 4; ++i)
        tile[ly + i * 8][lx] = f32_bf16(w[(size_t)(k0 + ly + i * 8) * N + n0 + lx]);
    __syncthreads();
#pragma unroll
    for (int i = 0; i < 4; ++i)
        out[(size_t)(n0 + ly + i * 8) * K + k0 + lx] = tile[lx][ly + i * 8];
}

// -------------------------- stage one 32-deep K-slab of A/B tiles into LDS buffer
__device__ __forceinline__ void stage_tiles(const unsigned short* __restrict__ A,
                                            const unsigned short* __restrict__ Bt,
                                            int Kdim, int m0, int n0, int k0,
                                            int tid,
                                            unsigned short* lAbuf,
                                            unsigned short* lBbuf) {
#pragma unroll
    for (int i = 0; i < 2; ++i) {               // A tile: 128x32 = 512 x 16B chunks
        int c = tid + (i << 8);
        int r = c >> 2, cc = (c & 3) << 3;
        cp16_g2l(&A[(size_t)(m0 + r) * Kdim + k0 + cc], &lAbuf[(r << 5) + cc]);
    }
#pragma unroll
    for (int i = 0; i < 4; ++i) {               // B tile: 256x32 = 1024 x 16B chunks
        int c = tid + (i << 8);
        int r = c >> 2, cc = (c & 3) << 3;
        cp16_g2l(&Bt[(size_t)(n0 + r) * Kdim + k0 + cc], &lBbuf[(r << 5) + cc]);
    }
}

// --------------------------------------------------------------- tiled bf16 GEMM
// C[M][N] = A[M][K] * Bt[N][K]^T ; block tile 128x256, 8 waves of 64x64.
// Double-buffered LDS, async global->LDS copies overlapped with WMMA.
// mode 0: scatter into q [BH][T][64] (x0.125), k [BH][T][64], vT [BH][64][T]
// mode 1: write fp32 to outf[M][N]
__global__ __launch_bounds__(256)
void k_gemm(const unsigned short* __restrict__ A,
            const unsigned short* __restrict__ Bt,
            int Ndim, int Kdim, int mode,
            unsigned short* __restrict__ qb,
            unsigned short* __restrict__ kbuf,
            unsigned short* __restrict__ vT,
            float* __restrict__ outf) {
    __shared__ __align__(16) unsigned short lA[2][128 * 32];   // 2 x 8 KB
    __shared__ __align__(16) unsigned short lB[2][256 * 32];   // 2 x 16 KB
    const int tid = threadIdx.x;
    const int wave = tid >> 5, lane = tid & 31;
    const int lrow = lane & 15, lhalf = lane >> 4, kb8 = lhalf << 3;
    const int nblk = Ndim >> 8;
    const int m0 = (int)(blockIdx.x / nblk) << 7;
    const int n0 = (int)(blockIdx.x % nblk) << 8;
    const int mw = (wave & 1) << 6;
    const int nw = (wave >> 1) << 6;
    const f32x8 vz = {0.f, 0.f, 0.f, 0.f, 0.f, 0.f, 0.f, 0.f};
    f32x8 acc[4][4];
#pragma unroll
    for (int ai = 0; ai < 4; ++ai)
#pragma unroll
        for (int bi = 0; bi < 4; ++bi) acc[ai][bi] = vz;

    const int nst = Kdim >> 5;
    stage_tiles(A, Bt, Kdim, m0, n0, 0, tid, lA[0], lB[0]);

    for (int s = 0; s < nst; ++s) {
        const int buf = s & 1;
        if (s + 1 < nst) {
            stage_tiles(A, Bt, Kdim, m0, n0, (s + 1) << 5, tid, lA[buf ^ 1], lB[buf ^ 1]);
            WAIT_ASYNC(6);       // stage s's 6 per-thread copies have landed (in-order)
        } else {
            WAIT_ASYNC(0);
        }
        __syncthreads();

        bf16x16 af[4], bfr[4];
#pragma unroll
        for (int ai = 0; ai < 4; ++ai) {
            const unsigned short* p = &lA[buf][(mw + (ai << 4) + lrow) << 5];
            af[ai] = load_frag(p + kb8, p + 16 + kb8);
        }
#pragma unroll
        for (int bi = 0; bi < 4; ++bi) {
            const unsigned short* p = &lB[buf][(nw + (bi << 4) + lrow) << 5];
            bfr[bi] = load_frag(p + kb8, p + 16 + kb8);
        }
#pragma unroll
        for (int ai = 0; ai < 4; ++ai)
#pragma unroll
            for (int bi = 0; bi < 4; ++bi)
                acc[ai][bi] = wmma_bf16(af[ai], bfr[bi], acc[ai][bi]);
        __syncthreads();         // all waves done reading buf before it is refilled
    }

#pragma unroll
    for (int ai = 0; ai < 4; ++ai)
#pragma unroll
        for (int bi = 0; bi < 4; ++bi)
#pragma unroll
            for (int r = 0; r < 8; ++r) {
                int gm = m0 + mw + (ai << 4) + r + (lhalf << 3);
                int gn = n0 + nw + (bi << 4) + lrow;
                float val = acc[ai][bi][r];
                if (mode == 0) {
                    int which = gn >> 10;            // 0=q 1=k 2=v
                    int dD = gn & 1023;
                    int h = dD >> 6, hd = dD & 63;
                    int bidx = gm >> 11, t = gm & (T_ - 1);
                    size_t bh = (size_t)bidx * H_ + h;
                    if (which == 0)
                        qb[((bh * T_ + t) << 6) + hd] = f32_bf16(val * 0.125f);
                    else if (which == 1)
                        kbuf[((bh * T_ + t) << 6) + hd] = f32_bf16(val);
                    else
                        vT[((bh << 6) + hd) * T_ + t] = f32_bf16(val);
                } else {
                    outf[(size_t)gm * Ndim + gn] = val;
                }
            }
}

// --------------------------------------------------------- flash attention (bf16)
// One wave owns 16 query rows; online softmax over 32-key steps.
__global__ __launch_bounds__(256)
void k_flash(const unsigned short* __restrict__ qb,
             const unsigned short* __restrict__ kbuf,
             const unsigned short* __restrict__ vT,
             unsigned short* __restrict__ yb) {
    __shared__ __align__(16) unsigned short pbuf[8][16 * 32]; // per-wave P staging
    const int tid = threadIdx.x, wave = tid >> 5, lane = tid & 31;
    const int lrow = lane & 15, lhalf = lane >> 4, kb8 = lhalf << 3;
    const int qt = blockIdx.x & 15;          // T_/128 = 16 q-tiles
    const int bh = blockIdx.x >> 4;
    const int qbase = (qt << 7) + (wave << 4);

    const unsigned short* qrow =
        qb + (((size_t)bh * T_ + qbase) << 6) + ((size_t)lrow << 6);
    bf16x16 qf0 = load_frag(qrow + kb8, qrow + 16 + kb8);       // hd 0..31
    bf16x16 qf1 = load_frag(qrow + 32 + kb8, qrow + 48 + kb8);  // hd 32..63

    const unsigned short* kbh = kbuf + ((size_t)bh * T_ << 6);
    const unsigned short* vbh = vT + (((size_t)bh << 6) * T_);
    unsigned short* pb = &pbuf[wave][0];

    const f32x8 vz = {0.f, 0.f, 0.f, 0.f, 0.f, 0.f, 0.f, 0.f};
    f32x8 o[4];
#pragma unroll
    for (int nt = 0; nt < 4; ++nt) o[nt] = vz;
    float mrun[8], lrun[8];
#pragma unroll
    for (int r = 0; r < 8; ++r) { mrun[r] = -1e30f; lrun[r] = 0.f; }

    const int qlast = qbase + 15;
    for (int kt = 0; kt <= qlast; kt += 32) {
        // K fragments: B-matrix, N=key, contract=hd (rows contiguous in memory)
        const unsigned short* kr0 = kbh + ((size_t)(kt + lrow) << 6);
        const unsigned short* kr1 = kbh + ((size_t)(kt + 16 + lrow) << 6);
        bf16x16 k00 = load_frag(kr0 + kb8, kr0 + 16 + kb8);
        bf16x16 k01 = load_frag(kr0 + 32 + kb8, kr0 + 48 + kb8);
        bf16x16 k10 = load_frag(kr1 + kb8, kr1 + 16 + kb8);
        bf16x16 k11 = load_frag(kr1 + 32 + kb8, kr1 + 48 + kb8);

        f32x8 s0 = vz, s1 = vz;
        s0 = wmma_bf16(qf0, k00, s0);  s0 = wmma_bf16(qf1, k01, s0);
        s1 = wmma_bf16(qf0, k10, s1);  s1 = wmma_bf16(qf1, k11, s1);

        float alpha[8];
#pragma unroll
        for (int r = 0; r < 8; ++r) {
            int qg = qbase + r + (lhalf << 3);
            float v0 = (kt + lrow <= qg) ? s0[r] : -1e30f;        // causal mask
            float v1 = (kt + 16 + lrow <= qg) ? s1[r] : -1e30f;
            float tm = fmaxf(v0, v1);
#pragma unroll
            for (int off = 1; off < 16; off <<= 1)
                tm = fmaxf(tm, __shfl_xor(tm, off, 32));          // 16-lane row max
            float mn = fmaxf(mrun[r], tm);
            float al = __expf(mrun[r] - mn);
            float p0 = __expf(v0 - mn);
            float p1 = __expf(v1 - mn);
            float ps = p0 + p1;
#pragma unroll
            for (int off = 1; off < 16; off <<= 1)
                ps += __shfl_xor(ps, off, 32);                    // 16-lane row sum
            lrun[r] = lrun[r] * al + ps;
            mrun[r] = mn;
            alpha[r] = al;
            int m = r + (lhalf << 3);
            pb[(m << 5) + lrow]      = f32_bf16(p0);   // P: [16 q][32 keys]
            pb[(m << 5) + 16 + lrow] = f32_bf16(p1);
        }
#pragma unroll
        for (int nt = 0; nt < 4; ++nt)
#pragma unroll
            for (int r = 0; r < 8; ++r) o[nt][r] *= alpha[r];

        asm volatile("s_wait_dscnt 0" ::: "memory");  // P stores visible in-wave

        const unsigned short* pr = pb + (lrow << 5);
        bf16x16 pf = load_frag(pr + kb8, pr + 16 + kb8);  // P as A-matrix 16x32
#pragma unroll
        for (int nt = 0; nt < 4; ++nt) {
            const unsigned short* vr =
                vbh + (size_t)((nt << 4) + lrow) * T_ + kt;
            bf16x16 vf = load_frag(vr + kb8, vr + 16 + kb8);  // V^T rows contiguous
            o[nt] = wmma_bf16(pf, vf, o[nt]);
        }
        asm volatile("" ::: "memory");  // keep next iter's DS stores after reads
    }

    const int bidx = bh >> 4, hh = bh & 15;
#pragma unroll
    for (int nt = 0; nt < 4; ++nt)
#pragma unroll
        for (int r = 0; r < 8; ++r) {
            int t = qbase + r + (lhalf << 3);
            int hd = (nt << 4) + lrow;
            yb[((size_t)bidx * T_ + t) * D_ + (hh << 6) + hd] =
                f32_bf16(o[nt][r] / lrun[r]);
        }
}

// -------------------------------------------------------------------------- host
extern "C" void kernel_launch(void* const* d_in, const int* in_sizes, int n_in,
                              void* d_out, int out_size, void* d_ws, size_t ws_size,
                              hipStream_t stream) {
    const float* x      = (const float*)d_in[0];
    const float* w_qkv  = (const float*)d_in[1];
    const float* w_proj = (const float*)d_in[2];
    float* out = (float*)d_out;

    char* ws = (char*)d_ws;
    size_t off = 0;
    auto take = [&](size_t bytes) {
        char* p = ws + off;
        off += (bytes + 255) & ~(size_t)255;
        return p;
    };
    unsigned short* xb     = (unsigned short*)take((size_t)M_ * D_ * 2);
    unsigned short* wqkvT  = (unsigned short*)take((size_t)N_QKV * D_ * 2);
    unsigned short* wprojT = (unsigned short*)take((size_t)D_ * D_ * 2);
    unsigned short* qb     = (unsigned short*)take((size_t)BH_ * T_ * HD_ * 2);
    unsigned short* kb     = (unsigned short*)take((size_t)BH_ * T_ * HD_ * 2);
    unsigned short* vT     = (unsigned short*)take((size_t)BH_ * T_ * HD_ * 2);
    unsigned short* yb     = (unsigned short*)take((size_t)M_ * D_ * 2);

    k_cast<<<(M_ * D_) / (256 * 4), 256, 0, stream>>>(x, xb, M_ * D_);
    k_transpose_cast<<<(N_QKV / 32) * (D_ / 32), 256, 0, stream>>>(w_qkv, wqkvT, D_, N_QKV);
    k_transpose_cast<<<(D_ / 32) * (D_ / 32), 256, 0, stream>>>(w_proj, wprojT, D_, D_);

    k_gemm<<<(M_ / 128) * (N_QKV / 256), 256, 0, stream>>>(
        xb, wqkvT, N_QKV, D_, 0, qb, kb, vT, nullptr);

    k_flash<<<BH_ * (T_ / 128), 256, 0, stream>>>(qb, kb, vT, yb);

    k_gemm<<<(M_ / 128) * (D_ / 256), 256, 0, stream>>>(
        yb, wprojT, D_, D_, 1, nullptr, nullptr, nullptr, out);
}
